// OSSModule_27513560498492
// MI455X (gfx1250) — matrix operand order repaired and checked by reading, not compiled, not run on hardware
//
#include <hip/hip_runtime.h>
#include <math.h>

// ---------------------------------------------------------------------------
// Types / helpers
// ---------------------------------------------------------------------------
typedef __attribute__((ext_vector_type(16))) __bf16 v16bf;
typedef __attribute__((ext_vector_type(8)))  float  v8f;
typedef unsigned short ush;

__device__ __forceinline__ ush f2bf(float f) {
  unsigned u = __builtin_bit_cast(unsigned, f);
  unsigned r = (u + 0x7FFFu + ((u >> 16) & 1u)) >> 16;
  return (ush)r;
}
__device__ __forceinline__ float silu(float x) { return x / (1.f + __expf(-x)); }
__device__ __forceinline__ float softplus(float x) {
  return x > 20.f ? x : log1pf(__expf(x));
}

#define BB 4
#define CC 64
#define HWN 4096
#define LL 4096
#define BL 16384          // B*L

// Async global->LDS staging (ASYNCcnt path), if the toolchain exposes it.
#if defined(__has_builtin)
#if __has_builtin(__builtin_amdgcn_global_load_async_to_lds_b128)
#define HAVE_ASYNC_LDS 1
#endif
#if __has_builtin(__builtin_amdgcn_s_wait_asynccnt)
#define HAVE_WAIT_ASYNC 1
#endif
#endif

#ifdef HAVE_ASYNC_LDS
typedef int v4i __attribute__((vector_size(16)));
typedef __attribute__((address_space(1))) v4i* gptr_v4i;  // global
typedef __attribute__((address_space(3))) v4i* lptr_v4i;  // LDS
__device__ __forceinline__ void async_copy16(const void* gsrc, void* ldst) {
  __builtin_amdgcn_global_load_async_to_lds_b128((gptr_v4i)(void*)gsrc,
                                                 (lptr_v4i)ldst, 0, 0);
}
__device__ __forceinline__ void async_wait0() {
#ifdef HAVE_WAIT_ASYNC
  __builtin_amdgcn_s_wait_asynccnt(0);
#else
  asm volatile("s_wait_asynccnt 0" ::: "memory");
#endif
}
#endif

// ---------------------------------------------------------------------------
// K0: convert weight matrix W[K][N] (f32) -> Wt[N][K] (bf16), for WMMA B-operand
// ---------------------------------------------------------------------------
__global__ void wt_convert(const float* __restrict__ W, ush* __restrict__ Wt,
                           int K, int N) {
  int i = blockIdx.x * 256 + threadIdx.x;
  if (i >= K * N) return;
  int n = i / K, k = i - n * K;
  Wt[(size_t)n * K + k] = f2bf(W[(size_t)k * N + n]);
}

// ---------------------------------------------------------------------------
// K1a: input pointwise projection  y[b,o,p] = sum_c x[b,c,p]*inp_w[o,c]+b[o]
// ---------------------------------------------------------------------------
__global__ __launch_bounds__(256)
void k1a_inproj(const float* __restrict__ x, const float* __restrict__ w,
                const float* __restrict__ bias, float* __restrict__ fo1pre,
                float* __restrict__ fo2) {
  int b = blockIdx.y, og = blockIdx.x, tid = threadIdx.x;
  __shared__ float wl[8][64];
  __shared__ float bl[8];
  for (int i = tid; i < 512; i += 256)
    wl[i >> 6][i & 63] = w[(og * 8 + (i >> 6)) * 64 + (i & 63)];
  if (tid < 8) bl[tid] = bias[og * 8 + tid];
  __syncthreads();
  for (int it = 0; it < 16; ++it) {
    int p = it * 256 + tid;
    float acc[8] = {0, 0, 0, 0, 0, 0, 0, 0};
    for (int c = 0; c < 64; ++c) {
      float v = x[((size_t)(b * 64 + c) << 12) + p];
#pragma unroll
      for (int j = 0; j < 8; ++j) acc[j] += v * wl[j][c];
    }
#pragma unroll
    for (int j = 0; j < 8; ++j) {
      int o = og * 8 + j;
      float s = acc[j] + bl[j];
      if (o < 64)
        fo1pre[((size_t)(b * 64 + o) << 12) + p] = s;
      else
        fo2[((size_t)(b * 64 + (o - 64)) << 12) + p] = silu(s);
    }
  }
}

// ---------------------------------------------------------------------------
// K1b: depthwise 3x3 SAME conv + bias + SiLU, scatter 4 directional bf16 copies
// ---------------------------------------------------------------------------
__global__ __launch_bounds__(256)
void k1b_dwconv(const float* __restrict__ fo1pre, const float* __restrict__ dww,
                const float* __restrict__ dwb, ush* __restrict__ fop) {
  int idx = blockIdx.x * 256 + threadIdx.x;   // B*C*HW = 1M
  int w = idx & 63, h = (idx >> 6) & 63, c = (idx >> 12) & 63, b = idx >> 18;
  float s = dwb[c];
#pragma unroll
  for (int i = 0; i < 3; ++i)
#pragma unroll
    for (int j = 0; j < 3; ++j) {
      int hh = h + i - 1, ww = w + j - 1;
      if (hh >= 0 && hh < 64 && ww >= 0 && ww < 64)
        s += dww[c * 9 + i * 3 + j] *
             fo1pre[((size_t)(b * 64 + c) << 12) + (hh << 6) + ww];
    }
  ush bv = f2bf(silu(s));
  int lh = (h << 6) + w, lw = (w << 6) + h;
  size_t base = (size_t)b * LL;
  fop[(base + lh) * 256 + c] = bv;
  fop[(base + (LL - 1 - lh)) * 256 + 64 + c] = bv;
  fop[(base + lw) * 256 + 128 + c] = bv;
  fop[(base + (LL - 1 - lw)) * 256 + 192 + c] = bv;
}

// ---------------------------------------------------------------------------
// Generic bf16 WMMA GEMM:  C[M][N](f32) = A[M][K](bf16) * B[K][N]
// B supplied transposed: Bt[N][K] (bf16).
// Block = 8 waves, tile 64(M) x 128(N): each wave holds 4 accumulators and
// reuses its B fragment across 4 WMMAs per K-step.
// grid = (M/64, ceil(N/128))
// ---------------------------------------------------------------------------
__global__ __launch_bounds__(256)
void gemm_bf16(const ush* __restrict__ A, const ush* __restrict__ Bt,
               float* __restrict__ Cm, int M, int N, int K) {
  __shared__ ush lA[64 * 32];
  __shared__ ush lB[128 * 32];
  int tid = threadIdx.x;
  int wave = tid >> 5, lane = tid & 31;
  int m0 = blockIdx.x * 64;
  int n0 = blockIdx.y * 128;
  v8f acc[4] = {v8f{}, v8f{}, v8f{}, v8f{}};
  int arow = tid >> 2, aseg = (tid & 3) * 8;   // A: 64 rows x 32, 16B/thread
  int brow = tid >> 1, bseg = (tid & 1) * 16;  // B: 128 rows x 32, 32B/thread

  for (int k0 = 0; k0 < K; k0 += 32) {
#ifdef HAVE_ASYNC_LDS
    async_copy16(&A[(size_t)(m0 + arow) * K + k0 + aseg], &lA[arow * 32 + aseg]);
    if (n0 + brow < N) {
      async_copy16(&Bt[(size_t)(n0 + brow) * K + k0 + bseg],
                   &lB[brow * 32 + bseg]);
      async_copy16(&Bt[(size_t)(n0 + brow) * K + k0 + bseg + 8],
                   &lB[brow * 32 + bseg + 8]);
    }
    async_wait0();
#else
    *(uint4*)&lA[arow * 32 + aseg] =
        *(const uint4*)&A[(size_t)(m0 + arow) * K + k0 + aseg];
    if (n0 + brow < N) {
      const ush* src = &Bt[(size_t)(n0 + brow) * K + k0 + bseg];
      ush* dst = &lB[brow * 32 + bseg];
      ((uint4*)dst)[0] = ((const uint4*)src)[0];
      ((uint4*)dst)[1] = ((const uint4*)src)[1];
    }
#endif
    __syncthreads();
    // Fragment builds per ISA VGPR-layout tables (wave32).
    int mrow = lane & 15;
    int kb8 = (lane >> 4) * 8;    // A: two 8-elem runs split at +16
    int kb16 = (lane >> 4) * 16;  // unused; B uses same split-run K order as A
    int ncol = wave * 16 + (lane & 15);
    const __bf16* pa = (const __bf16*)lA;
    const __bf16* pb = (const __bf16*)lB;
    v16bf bfrag;
#pragma unroll
    for (int j = 0; j < 8; ++j) {
      bfrag[j]     = pb[ncol * 32 + kb8 + j];
      bfrag[8 + j] = pb[ncol * 32 + kb8 + 16 + j];
    }
#pragma unroll
    for (int mt = 0; mt < 4; ++mt) {
      v16bf afrag;
#pragma unroll
      for (int j = 0; j < 8; ++j) {
        afrag[j]     = pa[(mt * 16 + mrow) * 32 + kb8 + j];
        afrag[8 + j] = pa[(mt * 16 + mrow) * 32 + kb8 + 16 + j];
      }
      acc[mt] = __builtin_amdgcn_wmma_f32_16x16x32_bf16(
          false, afrag, false, bfrag, (short)0, acc[mt], false, false);
    }
    __syncthreads();
  }
  int ncol = n0 + wave * 16 + (lane & 15);
  if (ncol < N) {
#pragma unroll
    for (int mt = 0; mt < 4; ++mt) {
      int rbase = m0 + mt * 16 + ((lane >> 4) * 8);
#pragma unroll
      for (int r = 0; r < 8; ++r)
        Cm[(size_t)(rbase + r) * N + ncol] = acc[mt][r];
    }
  }
}

// ---------------------------------------------------------------------------
// K3: causal depthwise conv (K=4) along L on xs = xz[:, :512], + bias + SiLU
// ---------------------------------------------------------------------------
__global__ __launch_bounds__(256)
void k3_conv1d(const float* __restrict__ xz, const float* __restrict__ cw,
               const float* __restrict__ cb, float* __restrict__ xsc,
               ush* __restrict__ xscbf) {
  int idx = blockIdx.x * 256 + threadIdx.x;  // B*L*512
  int d = idx & 511, l = (idx >> 9) & 4095, b = idx >> 21;
  float s = cb[d];
#pragma unroll
  for (int k = 0; k < 4; ++k) {
    int ll = l - 3 + k;
    if (ll >= 0) s += cw[d * 4 + k] * xz[(size_t)(b * LL + ll) * 1024 + d];
  }
  float u = silu(s);
  size_t row = (size_t)(b * LL + l);
  xsc[row * 512 + d] = u;
  xscbf[row * 512 + d] = f2bf(u);
}

// ---------------------------------------------------------------------------
// K5: dt = softplus(dbc[:, :16] @ dt_w + dt_b)
// ---------------------------------------------------------------------------
__global__ __launch_bounds__(256)
void k5_dt(const float* __restrict__ dbc, const float* __restrict__ dtw,
           const float* __restrict__ dtb, float* __restrict__ dt) {
  int idx = blockIdx.x * 256 + threadIdx.x;  // B*L*512
  int d = idx & 511;
  size_t row = (size_t)(idx >> 9);
  float s = dtb[d];
#pragma unroll
  for (int r = 0; r < 16; ++r) s += dbc[row * 48 + r] * dtw[r * 512 + d];
  dt[row * 512 + d] = softplus(s);
}

// ---------------------------------------------------------------------------
// K6: selective scan.  One thread per (b, d, n); 16-lane shuffle reduce over n.
// ---------------------------------------------------------------------------
__global__ __launch_bounds__(256)
void k6_scan(const float* __restrict__ dt, const float* __restrict__ dbc,
             const float* __restrict__ u, const float* __restrict__ xz,
             const float* __restrict__ alog, const float* __restrict__ Dp,
             ush* __restrict__ ygate) {
  int t = blockIdx.x * 256 + threadIdx.x;  // 4*512*16 = 32768
  int n = t & 15, d = (t >> 4) & 511, b = t >> 13;
  float a = -__expf(alog[d * 16 + n]);
  float Dd = Dp[d];
  float h = 0.f;
  for (int l = 0; l < 4096; ++l) {
    size_t row = (size_t)(b * LL + l);
    if (l + 8 < 4096) {  // software prefetch (global_prefetch_b8)
      __builtin_prefetch(&dt[(row + 8) * 512 + d], 0, 1);
      __builtin_prefetch(&u[(row + 8) * 512 + d], 0, 1);
    }
    float dtv = dt[row * 512 + d];
    float uv = u[row * 512 + d];
    float Bn = dbc[row * 48 + 16 + n];
    float Cn = dbc[row * 48 + 32 + n];
    h = __expf(dtv * a) * h + dtv * Bn * uv;
    float part = h * Cn;
    part += __shfl_xor(part, 1, 16);
    part += __shfl_xor(part, 2, 16);
    part += __shfl_xor(part, 4, 16);
    part += __shfl_xor(part, 8, 16);
    if (n == 0) {
      float zv = xz[row * 1024 + 512 + d];
      float y = (part + uv * Dd) * silu(zv);
      ygate[row * 512 + d] = f2bf(y);
    }
  }
}

// ---------------------------------------------------------------------------
// K8: dir-sum + residual = fop_sum * fo2 ; per-(b,c) mean over HW
// ---------------------------------------------------------------------------
__global__ __launch_bounds__(256)
void k8_residual(const float* __restrict__ ym1, const float* __restrict__ fo2,
                 float* __restrict__ res, float* __restrict__ mean) {
  int bc = blockIdx.x;  // 0..255
  int b = bc >> 6, c = bc & 63;
  float s = 0.f;
  for (int it = 0; it < 16; ++it) {
    int p = it * 256 + threadIdx.x;
    size_t row = (size_t)(b * LL + p) * 256;
    float v = ym1[row + c] + ym1[row + 64 + c] + ym1[row + 128 + c] +
              ym1[row + 192 + c];
    float r = v * fo2[((size_t)bc << 12) + p];
    res[((size_t)bc << 12) + p] = r;
    s += r;
  }
  __shared__ float red[256];
  red[threadIdx.x] = s;
  __syncthreads();
  for (int o = 128; o; o >>= 1) {
    if (threadIdx.x < o) red[threadIdx.x] += red[threadIdx.x + o];
    __syncthreads();
  }
  if (threadIdx.x == 0) mean[bc] = red[0] * (1.f / 4096.f);
}

// ---------------------------------------------------------------------------
// K9: tiny channel mamba (L=64, d=2, di=4, ds=16). One thread per batch.
// ---------------------------------------------------------------------------
__global__ void k9_mamba2(const float* __restrict__ mean,
                          const float* __restrict__ inw,
                          const float* __restrict__ cw,
                          const float* __restrict__ cb,
                          const float* __restrict__ xpw,
                          const float* __restrict__ dtw,
                          const float* __restrict__ dtb,
                          const float* __restrict__ alog,
                          const float* __restrict__ Dp,
                          const float* __restrict__ outw,
                          float* __restrict__ foc) {
  int b = threadIdx.x;
  if (b >= 4) return;
  float h[4][16], A2[4][16], hist[4][4];
  for (int d = 0; d < 4; ++d)
    for (int n = 0; n < 16; ++n) {
      h[d][n] = 0.f;
      A2[d][n] = -__expf(alog[d * 16 + n]);
    }
  for (int d = 0; d < 4; ++d)
    for (int k = 0; k < 4; ++k) hist[d][k] = 0.f;
  for (int l = 0; l < 64; ++l) {
    float in0 = mean[b * 64 + l], in1 = mean[b * 64 + 63 - l];
    float xr[4], zv[4], u[4];
    for (int j = 0; j < 4; ++j) {
      xr[j] = in0 * inw[j] + in1 * inw[8 + j];
      zv[j] = in0 * inw[4 + j] + in1 * inw[12 + j];
    }
    for (int d = 0; d < 4; ++d) {
      hist[d][0] = hist[d][1]; hist[d][1] = hist[d][2];
      hist[d][2] = hist[d][3]; hist[d][3] = xr[d];
      float s = cb[d];
      for (int k = 0; k < 4; ++k) s += cw[d * 4 + k] * hist[d][k];
      u[d] = silu(s);
    }
    float dtr = 0.f, Bv[16], Cv[16];
    for (int n = 0; n < 16; ++n) { Bv[n] = 0.f; Cv[n] = 0.f; }
    for (int d = 0; d < 4; ++d) {
      dtr += u[d] * xpw[d * 33];
      for (int n = 0; n < 16; ++n) {
        Bv[n] += u[d] * xpw[d * 33 + 1 + n];
        Cv[n] += u[d] * xpw[d * 33 + 17 + n];
      }
    }
    float o0 = 0.f, o1 = 0.f;
    for (int d = 0; d < 4; ++d) {
      float dtv = softplus(dtr * dtw[d] + dtb[d]);
      float yd = 0.f;
      for (int n = 0; n < 16; ++n) {
        h[d][n] = __expf(dtv * A2[d][n]) * h[d][n] + dtv * Bv[n] * u[d];
        yd += h[d][n] * Cv[n];
      }
      yd = (yd + u[d] * Dp[d]) * silu(zv[d]);
      o0 += yd * outw[d * 2 + 0];
      o1 += yd * outw[d * 2 + 1];
    }
    foc[b * 64 + l] = o0 + o1;
  }
}

// ---------------------------------------------------------------------------
// K10: foss = res*(1+foc) ; out = outp_w @ foss + outp_b.  LDS-staged W.
// ---------------------------------------------------------------------------
__global__ __launch_bounds__(256)
void k10_outproj(const float* __restrict__ res, const float* __restrict__ foc,
                 const float* __restrict__ W, const float* __restrict__ bias,
                 float* __restrict__ out) {
  __shared__ float wl[64 * 64];
  __shared__ float bl[64];
  int tid = threadIdx.x;
  for (int i = tid; i < 4096; i += 256) wl[i] = W[i];
  if (tid < 64) bl[tid] = bias[tid];
  __syncthreads();
  int gp = blockIdx.x * 256 + tid;  // 0..16383
  int b = gp >> 12, p = gp & 4095;
  float acc[64];
#pragma unroll
  for (int o = 0; o < 64; ++o) acc[o] = bl[o];
  for (int c = 0; c < 64; ++c) {
    float v = res[((size_t)(b * 64 + c) << 12) + p] * (1.f + foc[b * 64 + c]);
#pragma unroll
    for (int o = 0; o < 64; ++o) acc[o] += v * wl[o * 64 + c];
  }
#pragma unroll
  for (int o = 0; o < 64; ++o)
    out[((size_t)(b * 64 + o) << 12) + p] = acc[o];
}

// ---------------------------------------------------------------------------
// Host launcher
// ---------------------------------------------------------------------------
extern "C" void kernel_launch(void* const* d_in, const int* in_sizes, int n_in,
                              void* d_out, int out_size, void* d_ws,
                              size_t ws_size, hipStream_t stream) {
  const float* x        = (const float*)d_in[0];
  const float* inp_w    = (const float*)d_in[1];
  const float* inp_b    = (const float*)d_in[2];
  const float* dw_w     = (const float*)d_in[3];
  const float* dw_b     = (const float*)d_in[4];
  const float* outp_w   = (const float*)d_in[5];
  const float* outp_b   = (const float*)d_in[6];
  const float* m1_in_w  = (const float*)d_in[7];
  const float* m1_cw    = (const float*)d_in[8];
  const float* m1_cb    = (const float*)d_in[9];
  const float* m1_xpw   = (const float*)d_in[10];
  const float* m1_dtw   = (const float*)d_in[11];
  const float* m1_dtb   = (const float*)d_in[12];
  const float* m1_Alog  = (const float*)d_in[13];
  const float* m1_D     = (const float*)d_in[14];
  const float* m1_outw  = (const float*)d_in[15];
  const float* m2_in_w  = (const float*)d_in[16];
  const float* m2_cw    = (const float*)d_in[17];
  const float* m2_cb    = (const float*)d_in[18];
  const float* m2_xpw   = (const float*)d_in[19];
  const float* m2_dtw   = (const float*)d_in[20];
  const float* m2_dtb   = (const float*)d_in[21];
  const float* m2_Alog  = (const float*)d_in[22];
  const float* m2_D     = (const float*)d_in[23];
  const float* m2_outw  = (const float*)d_in[24];
  float* out = (float*)d_out;

  char* ws = (char*)d_ws;
  size_t off = 0;
  auto take = [&](size_t bytes) {
    size_t r = off;
    off += (bytes + 255) & ~(size_t)255;
    return r;
  };
  size_t oWtIn   = take((size_t)1024 * 256 * 2);
  size_t oWtXp   = take((size_t)48 * 512 * 2);
  size_t oWtOut  = take((size_t)256 * 512 * 2);
  size_t oFo1pre = take((size_t)BB * CC * HWN * 4);
  size_t oFo2    = take((size_t)BB * CC * HWN * 4);
  size_t oFopYg  = take((size_t)BL * 512 * 2);   // fop bf16, later ygate bf16
  size_t oXz     = take((size_t)BL * 1024 * 4);  // later reused as ym1 f32
  size_t oXsc    = take((size_t)BL * 512 * 4);
  size_t oXscBf  = take((size_t)BL * 512 * 2);
  size_t oDbc    = take((size_t)BL * 48 * 4);
  size_t oDt     = take((size_t)BL * 512 * 4);
  size_t oRes    = take((size_t)BB * CC * HWN * 4);
  size_t oMean   = take(256 * 4);
  size_t oFoc    = take(256 * 4);
  (void)ws_size; (void)n_in; (void)in_sizes; (void)out_size;

  ush*   WtIn  = (ush*)(ws + oWtIn);
  ush*   WtXp  = (ush*)(ws + oWtXp);
  ush*   WtOut = (ush*)(ws + oWtOut);
  float* fo1p  = (float*)(ws + oFo1pre);
  float* fo2   = (float*)(ws + oFo2);
  ush*   fop   = (ush*)(ws + oFopYg);
  ush*   ygate = (ush*)(ws + oFopYg);
  float* xz    = (float*)(ws + oXz);
  float* ym1   = (float*)(ws + oXz);
  float* xsc   = (float*)(ws + oXsc);
  ush*   xscbf = (ush*)(ws + oXscBf);
  float* dbc   = (float*)(ws + oDbc);
  float* dt    = (float*)(ws + oDt);
  float* res   = (float*)(ws + oRes);
  float* mean  = (float*)(ws + oMean);
  float* foc   = (float*)(ws + oFoc);

  // weight conversion (to bf16, [N][K])
  wt_convert<<<(256 * 1024 + 255) / 256, 256, 0, stream>>>(m1_in_w, WtIn, 256, 1024);
  wt_convert<<<(512 * 48 + 255) / 256, 256, 0, stream>>>(m1_xpw, WtXp, 512, 48);
  wt_convert<<<(512 * 256 + 255) / 256, 256, 0, stream>>>(m1_outw, WtOut, 512, 256);

  // input projection + depthwise conv + 4-direction pack
  k1a_inproj<<<dim3(16, 4), 256, 0, stream>>>(x, inp_w, inp_b, fo1p, fo2);
  k1b_dwconv<<<(BB * CC * HWN) / 256, 256, 0, stream>>>(fo1p, dw_w, dw_b, fop);

  // GEMM1: xz = fop @ m1_in_w   (16384 x 1024 x 256)
  gemm_bf16<<<dim3(BL / 64, 1024 / 128), 256, 0, stream>>>(fop, WtIn, xz, BL, 1024, 256);

  // causal conv1d + silu
  k3_conv1d<<<(BL * 512) / 256, 256, 0, stream>>>(xz, m1_cw, m1_cb, xsc, xscbf);

  // GEMM2: dbc = xs @ xproj_w   (16384 x 48 x 512)
  gemm_bf16<<<dim3(BL / 64, 1), 256, 0, stream>>>(xscbf, WtXp, dbc, BL, 48, 512);

  // dt = softplus(...)
  k5_dt<<<(BL * 512) / 256, 256, 0, stream>>>(dbc, m1_dtw, m1_dtb, dt);

  // selective scan + gate (writes ygate over fop region; xz z-half still live)
  k6_scan<<<(BB * 512 * 16) / 256, 256, 0, stream>>>(dt, dbc, xsc, xz, m1_Alog,
                                                     m1_D, ygate);

  // GEMM3: ym1 = ygate @ m1_out_w (16384 x 256 x 512); ym1 overlays dead xz
  gemm_bf16<<<dim3(BL / 64, 256 / 128), 256, 0, stream>>>(ygate, WtOut, ym1, BL, 256, 512);

  // direction-sum, residual, channel means
  k8_residual<<<256, 256, 0, stream>>>(ym1, fo2, res, mean);

  // tiny channel mamba
  k9_mamba2<<<1, 32, 0, stream>>>(mean, m2_in_w, m2_cw, m2_cb, m2_xpw, m2_dtw,
                                  m2_dtb, m2_Alog, m2_D, m2_outw, foc);

  // final scale + output projection
  k10_outproj<<<BL / 256, 256, 0, stream>>>(res, foc, outp_w, outp_b, out);
}
